// LinearAttention_65755949302267
// MI455X (gfx1250) — matrix operand order repaired and checked by reading, not compiled, not run on hardware
//
#include <hip/hip_runtime.h>
#include <hip/hip_bf16.h>
#include <cstdint>
#include <cstring>

typedef __bf16 bf16;
typedef __attribute__((ext_vector_type(16))) __bf16 v16bf;
typedef __attribute__((ext_vector_type(8)))  float  v8f;

#define DEV __device__ __forceinline__

// ---- sizes (fixed by the reference) ----
// B=2, N=8192, C=512, H=8, D=64, chunk L=64, chunks/seq=128, BH=16, rows M=16384

DEV bf16 f2bf(float f) {
  unsigned u = __builtin_bit_cast(unsigned, f);
  unsigned r = u + 0x7FFFu + ((u >> 16) & 1u);   // round-to-nearest-even
  unsigned short h = (unsigned short)(r >> 16);
  bf16 o; __builtin_memcpy(&o, &h, 2); return o;
}

DEV v8f wmma_bf16(v16bf a, v16bf b, v8f c) {
  // D = A(16x32) * B(32x16) + C, f32 accum
  return __builtin_amdgcn_wmma_f32_16x16x32_bf16(false, a, false, b, (short)0, c, false, false);
}

// A operand: 16x32 tile of row-major src[ld], rows row0.., K from k0.
// lane<16: row=lane, K elems {k0..k0+15}; lane>=16: {k0+16..k0+31}
DEV v16bf load_A(const bf16* src, int ld, int row0, int k0, int lane) {
  int m = lane & 15, kh = (lane >> 4) & 1;
  const bf16* p = src + (size_t)(row0 + m) * ld + k0 + kh * 16;
  v16bf a;
#pragma unroll
  for (int i = 0; i < 16; ++i) a[i] = p[i];
  return a;
}

// B operand where logical B = X^T, X row-major [rows, K]: B[k][n] = X[col0+n][k]
// (contiguous 16 bf16 per lane)
DEV v16bf load_BT(const bf16* X, int ld, int col0, int k0, int lane) {
  int n = lane & 15, kh = (lane >> 4) & 1;
  const bf16* p = X + (size_t)(col0 + n) * ld + k0 + kh * 16;
  v16bf b;
#pragma unroll
  for (int i = 0; i < 16; ++i) b[i] = p[i];
  return b;
}

// Same but source is fp32 (KV prefix state), converted on load
DEV v16bf load_BT_f32(const float* X, int ld, int col0, int k0, int lane) {
  int n = lane & 15, kh = (lane >> 4) & 1;
  const float* p = X + (size_t)(col0 + n) * ld + k0 + kh * 16;
  v16bf b;
#pragma unroll
  for (int i = 0; i < 16; ++i) b[i] = f2bf(p[i]);
  return b;
}

// B operand, row-major source: B[k][n] = X[k0+..][col0+n] (strided per lane)
DEV v16bf load_B_rm(const bf16* X, int ld, int k0, int col0, int lane) {
  int n = lane & 15, kh = (lane >> 4) & 1;
  const bf16* p = X + (size_t)(k0 + kh * 16) * ld + col0 + n;
  v16bf b;
#pragma unroll
  for (int i = 0; i < 16; ++i) b[i] = p[(size_t)i * ld];
  return b;
}

// ---------------- K0: fp32 -> bf16 conversions ----------------
__global__ void k_convert(const float* __restrict__ x, const float* __restrict__ wqkv,
                          const float* __restrict__ wproj,
                          bf16* __restrict__ xb, bf16* __restrict__ wqkvb,
                          bf16* __restrict__ wprojb) {
  int idx = blockIdx.x * blockDim.x + threadIdx.x;
  if (idx < 16384 * 512) xb[idx] = f2bf(x[idx]);
  if (idx < 1536 * 512)  wqkvb[idx] = f2bf(wqkv[idx]);
  if (idx < 512 * 512)   wprojb[idx] = f2bf(wproj[idx]);
}

// ---------------- K1: QKV GEMM + elu feature, scatter to Q/K/Vt ----------------
// grid (128, 24), block 256 (8 waves); wave tile 16(M) x 64(N); Kdim=512
__global__ void k_qkv(const bf16* __restrict__ xb, const bf16* __restrict__ wb,
                      bf16* __restrict__ Q, bf16* __restrict__ Kf, bf16* __restrict__ Vt) {
  const int lane = threadIdx.x & 31;
  const int wv   = threadIdx.x >> 5;            // 0..7
  const int row0 = blockIdx.x * 128 + wv * 16;  // row in [0,16384) = b*8192+n
  const int col0 = blockIdx.y * 64;             // col in [0,1536) = which*512 + h*64 + d
  v8f acc[4];
#pragma unroll
  for (int t = 0; t < 4; ++t) acc[t] = 0.0f;

  for (int k0 = 0; k0 < 512; k0 += 32) {
    v16bf a = load_A(xb, 512, row0, k0, lane);
#pragma unroll
    for (int t = 0; t < 4; ++t)
      acc[t] = wmma_bf16(a, load_BT(wb, 512, col0 + t * 16, k0, lane), acc[t]);
  }

  const int which = col0 / 512;          // uniform per block (64 | 512)
  const int h     = (col0 % 512) / 64;   // uniform per block
  const int mhi   = (lane & 16) ? 8 : 0;
  const int nloc  = lane & 15;
#pragma unroll
  for (int t = 0; t < 4; ++t) {
#pragma unroll
    for (int j = 0; j < 8; ++j) {
      float v = acc[t][j];
      int row = row0 + j + mhi;
      int b = row >> 13, nn = row & 8191;
      int d = t * 16 + nloc;
      if (which == 0) {
        float f = v > 0.f ? v + 1.f : __expf(v);          // elu(v)+1
        Q[(((size_t)(b * 8 + h)) * 8192 + nn) * 64 + d] = f2bf(f);
      } else if (which == 1) {
        float f = v > 0.f ? v + 1.f : __expf(v);
        Kf[(((size_t)(b * 8 + h)) * 8192 + nn) * 64 + d] = f2bf(f);
      } else {
        int ch = nn >> 6, l = nn & 63;                    // V stored transposed per chunk
        Vt[((((size_t)(b * 8 + h)) * 128 + ch) * 64 + d) * 64 + l] = f2bf(v);
      }
    }
  }
}

// ---------------- K2: per-chunk KV^T state = sum_l v_l k_l^T ----------------
// grid 2048 (= bh*128+chunk), block 128 (4 waves); KVt[e][d]
__global__ void k_chunkkv(const bf16* __restrict__ Kf, const bf16* __restrict__ Vt,
                          float* __restrict__ KV) {
  const int lane = threadIdx.x & 31;
  const int wv   = threadIdx.x >> 5;   // 0..3
  const int cid  = blockIdx.x;
  const bf16* kb = Kf + (size_t)cid * 4096;  // K chunk [l][d]
  const bf16* vb = Vt + (size_t)cid * 4096;  // Vt chunk [e][l]
  v8f acc[4];
#pragma unroll
  for (int t = 0; t < 4; ++t) acc[t] = 0.0f;
  for (int k0 = 0; k0 < 64; k0 += 32) {
    v16bf a = load_A(vb, 64, wv * 16, k0, lane);   // rows = e, contraction = l
#pragma unroll
    for (int t = 0; t < 4; ++t)
      acc[t] = wmma_bf16(a, load_B_rm(kb, 64, k0, t * 16, lane), acc[t]);
  }
  float* out = KV + (size_t)cid * 4096;
  const int mhi = (lane & 16) ? 8 : 0, nloc = lane & 15;
#pragma unroll
  for (int t = 0; t < 4; ++t)
#pragma unroll
    for (int j = 0; j < 8; ++j)
      out[(size_t)(wv * 16 + j + mhi) * 64 + t * 16 + nloc] = acc[t][j];
}

// ---------------- K3: exclusive prefix-sum of KV states (in place) ----------------
// grid 16 (=bh), block 256; each thread owns 16 consecutive floats of the 4096 state
__global__ void k_prefix(float* __restrict__ KV) {
  int bh = blockIdx.x, tid = threadIdx.x;
  float run[16];
#pragma unroll
  for (int i = 0; i < 16; ++i) run[i] = 0.f;
  for (int c = 0; c < 128; ++c) {
    float* p = KV + ((size_t)bh * 128 + c) * 4096 + tid * 16;
    float tmp[16];
#pragma unroll
    for (int i = 0; i < 16; ++i) { tmp[i] = p[i]; p[i] = run[i]; run[i] += tmp[i]; }
  }
}

// ---------------- K4: out = (Q K^T .* causal) V + Q * KV_prefix ----------------
// grid 2048 (=bh*128+chunk), block 128 (4 waves); each wave owns a 16-row band
__global__ void k_attn(const bf16* __restrict__ Q, const bf16* __restrict__ Kf,
                       const bf16* __restrict__ Vt, const float* __restrict__ KV,
                       bf16* __restrict__ attn) {
  __shared__ bf16 slds[64 * 64];
  const int lane = threadIdx.x & 31;
  const int wv   = threadIdx.x >> 5;    // 0..3
  const int cid  = blockIdx.x;
  const int bh = cid >> 7, ch = cid & 127;
  const int b = bh >> 3, h = bh & 7;
  const bf16*  qb  = Q  + (size_t)cid * 4096;   // [l][d]
  const bf16*  kb  = Kf + (size_t)cid * 4096;   // [l][d]
  const bf16*  vb  = Vt + (size_t)cid * 4096;   // [e][l]
  const float* kvp = KV + (size_t)cid * 4096;   // KVt prefix [e][d]
  __builtin_prefetch(kvp + lane * 64, 0, 1);    // global_prefetch_b8
  const int li0 = wv * 16;
  const int mhi = (lane & 16) ? 8 : 0, nloc = lane & 15;

  // S band = Q_band * K^T   (contraction over d)
  v8f sacc[4];
#pragma unroll
  for (int t = 0; t < 4; ++t) sacc[t] = 0.0f;
  for (int k0 = 0; k0 < 64; k0 += 32) {
    v16bf a = load_A(qb, 64, li0, k0, lane);
#pragma unroll
    for (int t = 0; t < 4; ++t)
      sacc[t] = wmma_bf16(a, load_BT(kb, 64, t * 16, k0, lane), sacc[t]);
  }
  // causal mask (inclusive) + C-layout -> LDS (this wave's private rows only;
  // LDS ops from one wave are in-order, no barrier needed)
#pragma unroll
  for (int t = 0; t < 4; ++t)
#pragma unroll
    for (int j = 0; j < 8; ++j) {
      int li = li0 + j + mhi, lj = t * 16 + nloc;
      slds[li * 64 + lj] = f2bf(lj <= li ? sacc[t][j] : 0.f);
    }

  v8f oacc[4];
#pragma unroll
  for (int t = 0; t < 4; ++t) oacc[t] = 0.0f;
  // Q_band * KV_prefix  (contraction over d)
  for (int k0 = 0; k0 < 64; k0 += 32) {
    v16bf a = load_A(qb, 64, li0, k0, lane);
#pragma unroll
    for (int t = 0; t < 4; ++t)
      oacc[t] = wmma_bf16(a, load_BT_f32(kvp, 64, t * 16, k0, lane), oacc[t]);
  }
  // S_masked * V  (contraction over lj; A from LDS, B = Vt contiguous)
  for (int k0 = 0; k0 < 64; k0 += 32) {
    v16bf a = load_A(slds, 64, li0, k0, lane);
#pragma unroll
    for (int t = 0; t < 4; ++t)
      oacc[t] = wmma_bf16(a, load_BT(vb, 64, t * 16, k0, lane), oacc[t]);
  }
  // store to attn [b, n, h*64+e] (row-major [16384,512] for the proj GEMM)
#pragma unroll
  for (int t = 0; t < 4; ++t)
#pragma unroll
    for (int j = 0; j < 8; ++j) {
      int li = li0 + j + mhi, e = t * 16 + nloc;
      int n = ch * 64 + li;
      attn[((size_t)b * 8192 + n) * 512 + h * 64 + e] = f2bf(oacc[t][j]);
    }
}

// ---------------- K5: projection GEMM + bias, fp32 out ----------------
// grid (128, 8), block 256 (8 waves); wave tile 16 x 64; Kdim=512
__global__ void k_proj(const bf16* __restrict__ attn, const bf16* __restrict__ wb,
                       const float* __restrict__ bias, float* __restrict__ out) {
  const int lane = threadIdx.x & 31;
  const int wv   = threadIdx.x >> 5;
  const int row0 = blockIdx.x * 128 + wv * 16;
  const int col0 = blockIdx.y * 64;
  v8f acc[4];
#pragma unroll
  for (int t = 0; t < 4; ++t) acc[t] = 0.0f;
  for (int k0 = 0; k0 < 512; k0 += 32) {
    v16bf a = load_A(attn, 512, row0, k0, lane);
#pragma unroll
    for (int t = 0; t < 4; ++t)
      acc[t] = wmma_bf16(a, load_BT(wb, 512, col0 + t * 16, k0, lane), acc[t]);
  }
  const int mhi = (lane & 16) ? 8 : 0, nloc = lane & 15;
#pragma unroll
  for (int t = 0; t < 4; ++t) {
    float bv = bias[col0 + t * 16 + nloc];
#pragma unroll
    for (int j = 0; j < 8; ++j) {
      int row = row0 + j + mhi;
      out[(size_t)row * 512 + col0 + t * 16 + nloc] = acc[t][j] + bv;
    }
  }
}

// ---------------- launcher ----------------
extern "C" void kernel_launch(void* const* d_in, const int* in_sizes, int n_in,
                              void* d_out, int out_size, void* d_ws, size_t ws_size,
                              hipStream_t stream) {
  const float* x     = (const float*)d_in[0];
  const float* wqkv  = (const float*)d_in[1];
  const float* wproj = (const float*)d_in[2];
  const float* bproj = (const float*)d_in[3];

  char* ws = (char*)d_ws;
  bf16*  xb     = (bf16*)(ws + 0);           // 16 MB
  bf16*  wqkvb  = (bf16*)(ws + 16777216);    // 1.5 MB
  bf16*  wprojb = (bf16*)(ws + 18350080);    // 0.5 MB
  bf16*  Q      = (bf16*)(ws + 18874368);    // 16 MB  [bh][n][d]
  bf16*  K      = (bf16*)(ws + 35651584);    // 16 MB  [bh][n][d]
  bf16*  Vt     = (bf16*)(ws + 52428800);    // 16 MB  [bh][ch][d][l]
  float* KV     = (float*)(ws + 69206016);   // 33.5 MB [bh][ch][e][d]
  bf16*  attn   = (bf16*)(ws + 102760448);   // 16 MB  [b][n][c]
  float* out    = (float*)d_out;

  k_convert<<<32768, 256, 0, stream>>>(x, wqkv, wproj, xb, wqkvb, wprojb);
  k_qkv    <<<dim3(128, 24), 256, 0, stream>>>(xb, wqkvb, Q, K, Vt);
  k_chunkkv<<<2048, 128, 0, stream>>>(K, Vt, KV);
  k_prefix <<<16, 256, 0, stream>>>(KV);
  k_attn   <<<2048, 128, 0, stream>>>(Q, K, Vt, KV, attn);
  k_proj   <<<dim3(128, 8), 256, 0, stream>>>(attn, wprojb, bproj, out);
}
